// CausalSelfAttention_36180804502115
// MI455X (gfx1250) — compile-verified, matrix operand order checked by reading, not run on hardware
//
#include <hip/hip_runtime.h>
#include <hip/hip_bf16.h>

// Causal self-attention for MI455X (gfx1250), bf16 WMMA path.
// B=4, T=2048, C=1024, H=16, D=64.

#define TT 2048
#define CCH 1024
#define NH 16
#define HD 64
#define BB 4

typedef __attribute__((ext_vector_type(16))) __bf16 v16bf;
typedef __attribute__((ext_vector_type(8)))  __bf16 v8bf;
typedef __attribute__((ext_vector_type(8)))  float  v8f;

// ---- WMMA wrapper: D = A(16x32 bf16) * B(32x16 bf16) + C(16x16 f32) ----
static __device__ __forceinline__ v8f wmma_bf16(v16bf a, v16bf b, v8f c) {
  return __builtin_amdgcn_wmma_f32_16x16x32_bf16(
      /*neg_a=*/false, a, /*neg_b=*/false, b,
      /*c_mod=*/(short)0, c, /*reuse_a=*/false, /*reuse_b=*/false);
}

// ---- Fragment loader for the "row x K" layout (A and transposed-B) ----
// ISA 7.12.2, 16-bit A 16x32: lane L: row = L&15, half h = L>>4;
// elements j=0..7  -> K = h*8 + j        (contiguous 16 bytes)
// elements j=8..15 -> K = 16 + h*8 + j-8 (contiguous 16 bytes)
static __device__ __forceinline__ v16bf
load_frag_rowk(const __bf16* __restrict__ base, int ld, int row, int k0, int lane) {
  const int m = lane & 15;
  const int h = lane >> 4;
  const __bf16* p = base + (size_t)(row + m) * ld + k0 + h * 8;
  v8bf lo = *(const v8bf*)(p);
  v8bf hi = *(const v8bf*)(p + 16);
  return __builtin_shufflevector(lo, hi, 0,1,2,3,4,5,6,7,8,9,10,11,12,13,14,15);
}

// ---- gfx1250 async copy: 16B per lane, global -> LDS (ASYNCcnt tracked) ----
static __device__ __forceinline__ void async_ld_b128(unsigned lds_off, const void* g) {
  asm volatile("global_load_async_to_lds_b128 %0, %1, off"
               :: "v"(lds_off), "v"(g) : "memory");
}
static __device__ __forceinline__ void wait_async0() {
  asm volatile("s_wait_asynccnt 0x0" ::: "memory");
}
// low 32 bits of a generic shared address == LDS byte offset on AMDGPU
static __device__ __forceinline__ unsigned lds_off_of(const void* p) {
  return (unsigned)(uintptr_t)p;
}

// ---- Prep kernels ----
__global__ void cvt_f32_bf16(const float* __restrict__ src, __bf16* __restrict__ dst, int n) {
  int i = blockIdx.x * blockDim.x + threadIdx.x;
  if (i < n) dst[i] = (__bf16)src[i];
}

// src [K,N] f32 -> dst [N,K] bf16
__global__ void transpose_f32_bf16(const float* __restrict__ src, __bf16* __restrict__ dst,
                                   int K, int N) {
  int i = blockIdx.x * blockDim.x + threadIdx.x;
  if (i < K * N) {
    int k = i / N, n = i - k * N;
    dst[(size_t)n * K + k] = (__bf16)src[i];
  }
}

// ---- QKV projection: [8192,1024] x [1024,3072] -> scatter Q/K/V^T ----
// Wave tile 32x64 (8 accumulators). Workgroup tile 128x128 (8 waves, 4m x 2n).
// Q: [B,H,T,D] (pre-scaled by 1/sqrt(D)), K: [B,H,T,D], Vt: [B,H,D,T]
__global__ __launch_bounds__(256)
void gemm_qkv(const __bf16* __restrict__ A,   // [8192,1024] row-major
              const __bf16* __restrict__ Bt,  // [3072,1024] row-major (W^T)
              const float* __restrict__ bias, // [3072]
              __bf16* __restrict__ Qb, __bf16* __restrict__ Kb, __bf16* __restrict__ Vt) {
  const int lane = threadIdx.x & 31;
  const int wave = threadIdx.x >> 5;
  const int m0 = blockIdx.x * 128 + (wave >> 1) * 32;
  const int n0 = blockIdx.y * 128 + (wave & 1)  * 64;

  v8f c[2][4] = {};
  for (int k0 = 0; k0 < CCH; k0 += 32) {
    v16bf a0 = load_frag_rowk(A, CCH, m0,      k0, lane);
    v16bf a1 = load_frag_rowk(A, CCH, m0 + 16, k0, lane);
    #pragma unroll
    for (int j = 0; j < 4; ++j) {
      v16bf bj = load_frag_rowk(Bt, CCH, n0 + j * 16, k0, lane);
      c[0][j] = wmma_bf16(a0, bj, c[0][j]);
      c[1][j] = wmma_bf16(a1, bj, c[1][j]);
    }
  }

  const int h = lane >> 4, nl = lane & 15;
  #pragma unroll
  for (int i = 0; i < 2; ++i)
    #pragma unroll
    for (int j = 0; j < 4; ++j)
      #pragma unroll
      for (int r = 0; r < 8; ++r) {
        int m = m0 + i * 16 + r + 8 * h;     // global row (b*T + t)
        int n = n0 + j * 16 + nl;            // global col in [0,3C)
        float v = c[i][j][r] + bias[n];
        int sec = n >> 10, cc = n & 1023;
        int hh = cc >> 6, dd = cc & 63;
        int b = m >> 11, t = m & 2047;
        size_t bh = (size_t)(b * NH + hh);
        if (sec == 0)       Qb[(bh * TT + t) * HD + dd] = (__bf16)(v * 0.125f);
        else if (sec == 1)  Kb[(bh * TT + t) * HD + dd] = (__bf16)v;
        else                Vt[(bh * HD + dd) * TT + t] = (__bf16)v;
      }
}

// ---- Flash attention: 4 waves x 16 query rows; 32-key blocks staged in LDS
//      via global_load_async_to_lds_b128; online softmax in registers. ----
__global__ __launch_bounds__(128)
void attention(const __bf16* __restrict__ Qb, const __bf16* __restrict__ Kb,
               const __bf16* __restrict__ Vt, __bf16* __restrict__ Y) {
  __shared__ __align__(16) __bf16 Ktile[32][HD];   // 32 keys x 64 d   (4 KB)
  __shared__ __align__(16) __bf16 Vtile[HD][32];   // 64 d    x 32 keys (4 KB)
  __shared__ __align__(16) __bf16 Pbuf[4][16][32]; // per-wave P repack (4 KB)

  const int tid  = threadIdx.x;
  const int lane = tid & 31;
  const int wave = tid >> 5;
  const int bh = blockIdx.y;                   // b*16 + h
  const int q0 = blockIdx.x * 64 + wave * 16;  // this wave's first query row
  const int q0wg = blockIdx.x * 64;            // workgroup's first query row

  const __bf16* Qh = Qb + (size_t)bh * TT * HD;
  const __bf16* Kh = Kb + (size_t)bh * TT * HD;
  const __bf16* Vh = Vt + (size_t)bh * HD * TT;

  // Q fragments for d=0..31 and d=32..63 (resident for whole kernel)
  const v16bf qa0 = load_frag_rowk(Qh, HD, q0, 0,  lane);
  const v16bf qa1 = load_frag_rowk(Qh, HD, q0, 32, lane);

  v8f o[4] = {};                 // O accumulator, 16q x 64d
  float rmax[8], rsum[8];        // lanes 0-15: rows q0+r ; lanes 16-31: rows q0+8+r
  #pragma unroll
  for (int r = 0; r < 8; ++r) { rmax[r] = -1e30f; rsum[r] = 0.0f; }

  const int h = lane >> 4, nl = lane & 15;
  const int kend    = ((q0   + 15) >> 5) << 5;  // this wave's last 32-key block
  const int kend_wg = ((q0wg + 63) >> 5) << 5;  // workgroup's last block

  for (int kb = 0; kb <= kend_wg; kb += 32) {
    __syncthreads();  // prior iteration's tile reads complete

    // Cooperative async DMA of K tile (contiguous 4 KB) and V^T tile into LDS.
    {
      const __bf16* gk = Kh + (size_t)kb * HD;   // rows kb..kb+31, contiguous
      #pragma unroll
      for (int u = 0; u < 2; ++u) {
        int ck = tid + u * 128;                  // 256 x 16B chunks
        async_ld_b128(lds_off_of((const char*)&Ktile[0][0] + ck * 16), gk + ck * 8);
        int d = ck >> 2, part = ck & 3;          // V^T: 64 rows x 4 chunks
        async_ld_b128(lds_off_of(&Vtile[d][part * 8]),
                      Vh + (size_t)d * TT + kb + part * 8);
      }
      wait_async0();
    }
    __syncthreads();  // tiles visible to all waves

    if (kb > kend) continue;   // fully masked for this wave's rows

    // S = Q K^T for 32 keys (two 16x16 C-fragments, K-dim = D = 64)
    v8f s0 = {}, s1 = {};
    s0 = wmma_bf16(qa0, load_frag_rowk(&Ktile[0][0],  HD, 0,  0,  lane), s0);
    s0 = wmma_bf16(qa1, load_frag_rowk(&Ktile[0][0],  HD, 0,  32, lane), s0);
    s1 = wmma_bf16(qa0, load_frag_rowk(&Ktile[0][0],  HD, 16, 0,  lane), s1);
    s1 = wmma_bf16(qa1, load_frag_rowk(&Ktile[0][0],  HD, 16, 32, lane), s1);

    const bool need_mask = (kb + 31) > q0;
    #pragma unroll
    for (int r = 0; r < 8; ++r) {
      const int qg = q0 + r + 8 * h;          // this element's global query row
      float v0 = s0[r], v1 = s1[r];
      if (need_mask) {
        if (kb + nl      > qg) v0 = -1e30f;
        if (kb + 16 + nl > qg) v1 = -1e30f;
      }
      // row max across the 16 lanes of this half
      float bm = fmaxf(v0, v1);
      #pragma unroll
      for (int off = 8; off >= 1; off >>= 1) bm = fmaxf(bm, __shfl_xor(bm, off, 16));
      const float nm = fmaxf(rmax[r], bm);
      const float sc = __expf(rmax[r] - nm);
      rmax[r] = nm;
      const float p0 = __expf(v0 - nm);
      const float p1 = __expf(v1 - nm);
      float ps = p0 + p1;
      #pragma unroll
      for (int off = 8; off >= 1; off >>= 1) ps += __shfl_xor(ps, off, 16);
      rsum[r] = rsum[r] * sc + ps;
      #pragma unroll
      for (int j = 0; j < 4; ++j) o[j][r] *= sc;   // same row mapping as S
      // stage P (C-layout -> LDS row-major 16x32); wave-private, DS in-order
      Pbuf[wave][r + 8 * h][nl]      = (__bf16)p0;
      Pbuf[wave][r + 8 * h][16 + nl] = (__bf16)p1;
    }

    // reload P as an A-fragment, then O += P (16x32) * V (32x64)
    const v16bf pf = load_frag_rowk(&Pbuf[wave][0][0], 32, 0, 0, lane);
    #pragma unroll
    for (int j = 0; j < 4; ++j) {
      v16bf vf = load_frag_rowk(&Vtile[0][0], 32, j * 16, 0, lane);
      o[j] = wmma_bf16(pf, vf, o[j]);
    }
  }

  // normalize and write Y in [B*T, C] bf16 layout for the output projection
  const int b = bh >> 4, hh = bh & 15;
  #pragma unroll
  for (int j = 0; j < 4; ++j)
    #pragma unroll
    for (int r = 0; r < 8; ++r) {
      const int m = q0 + r + 8 * h;
      Y[((size_t)b * TT + m) * CCH + hh * HD + j * 16 + nl] = (__bf16)(o[j][r] / rsum[r]);
    }
}

// ---- Output projection: Y[8192,1024] x Wp[1024,1024] + b -> f32 out ----
__global__ __launch_bounds__(256)
void gemm_proj(const __bf16* __restrict__ A,   // [8192,1024]
               const __bf16* __restrict__ Bt,  // [1024,1024] (W^T)
               const float* __restrict__ bias, // [1024]
               float* __restrict__ out) {
  const int lane = threadIdx.x & 31;
  const int wave = threadIdx.x >> 5;
  const int m0 = blockIdx.x * 128 + (wave >> 1) * 32;
  const int n0 = blockIdx.y * 128 + (wave & 1)  * 64;

  v8f c[2][4] = {};
  for (int k0 = 0; k0 < CCH; k0 += 32) {
    v16bf a0 = load_frag_rowk(A, CCH, m0,      k0, lane);
    v16bf a1 = load_frag_rowk(A, CCH, m0 + 16, k0, lane);
    #pragma unroll
    for (int j = 0; j < 4; ++j) {
      v16bf bj = load_frag_rowk(Bt, CCH, n0 + j * 16, k0, lane);
      c[0][j] = wmma_bf16(a0, bj, c[0][j]);
      c[1][j] = wmma_bf16(a1, bj, c[1][j]);
    }
  }
  const int h = lane >> 4, nl = lane & 15;
  #pragma unroll
  for (int i = 0; i < 2; ++i)
    #pragma unroll
    for (int j = 0; j < 4; ++j)
      #pragma unroll
      for (int r = 0; r < 8; ++r) {
        int m = m0 + i * 16 + r + 8 * h;
        int n = n0 + j * 16 + nl;
        out[(size_t)m * CCH + n] = c[i][j][r] + bias[n];
      }
}

extern "C" void kernel_launch(void* const* d_in, const int* in_sizes, int n_in,
                              void* d_out, int out_size, void* d_ws, size_t ws_size,
                              hipStream_t stream) {
  (void)in_sizes; (void)n_in; (void)out_size; (void)ws_size;
  const float* x      = (const float*)d_in[0];  // [4,2048,1024]
  const float* w_attn = (const float*)d_in[1];  // [1024,3072]
  const float* b_attn = (const float*)d_in[2];  // [3072]
  const float* w_proj = (const float*)d_in[3];  // [1024,1024]
  const float* b_proj = (const float*)d_in[4];  // [1024]
  float* out = (float*)d_out;                   // [4,2048,1024]

  char* ws = (char*)d_ws;
  size_t off = 0;
  auto alloc = [&](size_t bytes) { void* p = ws + off; off += (bytes + 255) & ~(size_t)255; return p; };

  const size_t MROWS = (size_t)BB * TT;         // 8192
  __bf16* xb     = (__bf16*)alloc(MROWS * CCH * 2);
  __bf16* wattnT = (__bf16*)alloc((size_t)3 * CCH * CCH * 2);
  __bf16* wprojT = (__bf16*)alloc((size_t)CCH * CCH * 2);
  __bf16* Qb     = (__bf16*)alloc(MROWS * CCH * 2);
  __bf16* Kb     = (__bf16*)alloc(MROWS * CCH * 2);
  __bf16* Vt     = (__bf16*)alloc(MROWS * CCH * 2);
  __bf16* Yb     = (__bf16*)alloc(MROWS * CCH * 2);

  const int nx = (int)(MROWS * CCH);
  cvt_f32_bf16<<<(nx + 255) / 256, 256, 0, stream>>>(x, xb, nx);
  transpose_f32_bf16<<<(CCH * 3 * CCH + 255) / 256, 256, 0, stream>>>(w_attn, wattnT, CCH, 3 * CCH);
  transpose_f32_bf16<<<(CCH * CCH + 255) / 256, 256, 0, stream>>>(w_proj, wprojT, CCH, CCH);

  gemm_qkv<<<dim3(64, 24), 256, 0, stream>>>(xb, wattnT, b_attn, Qb, Kb, Vt);
  attention<<<dim3(32, 64), 128, 0, stream>>>(Qb, Kb, Vt, Yb);
  gemm_proj<<<dim3(64, 8), 256, 0, stream>>>(Yb, wprojT, b_proj, out);
}
